// MaskedConv2d_71279277245044
// MI455X (gfx1250) — matrix operand order repaired and checked by reading, not compiled, not run on hardware
//
#include <hip/hip_runtime.h>
#include <stdint.h>

// ---------------------------------------------------------------------------
// Masked 3x3 conv (B=4, Cin=Cout=192, H=W=256, pad=1) as implicit GEMM on
// CDNA5 WMMA (v_wmma_f32_16x16x32_bf16), f32 emulated via bf16 hi/lo split
// (xh*wh + xh*wl + xl*wh, f32 accumulate, ~1e-5 rel err).
//
// Fast path: prepass converts w -> [tap][co][ci] and x -> NHWC bf16 hi/lo
// planes in d_ws; the main kernel stages LDS tiles with
// GLOBAL_LOAD_ASYNC_TO_LDS_B128 (ASYNCcnt-tracked, double buffered).
// Workgroup tile 64co x 128px; each wave computes 16co x 64px (4 subtiles)
// so one A-fragment pair feeds 12 WMMAs (20 ds_load_b128 per step).
// Fallback path (small ws): self-contained kernel converting in-loop.
// ---------------------------------------------------------------------------

typedef __attribute__((ext_vector_type(16))) __bf16    v16bf;
typedef __attribute__((ext_vector_type(8)))  float     v8f;
typedef __attribute__((ext_vector_type(4)))  uint32_t  u32x4;

#define BN    4
#define CIN   192
#define COUT  192
#define HD    256
#define WD    256

constexpr int MT  = 64;   // C_out tile per workgroup
constexpr int NTM = 128;  // pixels per workgroup (fast path)
constexpr int NT  = 64;   // pixels per workgroup (fallback)
constexpr int KC  = 32;   // ci chunk (K depth of one WMMA)
constexpr int RS  = 40;   // LDS row stride in bf16 elems (80B, 16B aligned)
constexpr int KSTEPS = 9 * (CIN / KC);   // 54

// workspace layout (bytes)
constexpr size_t WPLANE = (size_t)9 * COUT * CIN * 2;       //   663,552
constexpr size_t XPLANE = (size_t)BN * HD * WD * CIN * 2;   // 100,663,296
constexpr size_t WH_OFF = 0;
constexpr size_t WL_OFF = WH_OFF + WPLANE;
constexpr size_t XH_OFF = WL_OFF + WPLANE;
constexpr size_t XL_OFF = XH_OFF + XPLANE;
constexpr size_t WS_NEED = XL_OFF + XPLANE;

union Frag { v16bf v; u32x4 q[2]; };

__device__ __forceinline__ unsigned short f2bf(float f) {
    uint32_t u = __float_as_uint(f);
    u += 0x7FFFu + ((u >> 16) & 1u);          // round-to-nearest-even
    return (unsigned short)(u >> 16);
}
__device__ __forceinline__ float bf2f(unsigned short h) {
    return __uint_as_float(((uint32_t)h) << 16);
}

__device__ __forceinline__ float valid_at(const int* __restrict__ mask,
                                          int bb, int oh, int ow) {
    int any = 0;
#pragma unroll
    for (int dh = -1; dh <= 1; ++dh) {
        int hh = oh + dh;
        if (hh < 0 || hh >= HD) continue;
#pragma unroll
        for (int dw = -1; dw <= 1; ++dw) {
            int ww = ow + dw;
            if (ww < 0 || ww >= WD) continue;
            any |= (mask[(bb * HD + hh) * WD + ww] != 0);
        }
    }
    return any ? 1.0f : 0.0f;
}

// async 16B global -> LDS copy (ISA 10.x / 15.18.3, tracked with ASYNCcnt)
__device__ __forceinline__ void async_cp16(uint32_t lds_off, const void* g) {
    asm volatile("global_load_async_to_lds_b128 %0, %1, off"
                 :: "v"(lds_off), "v"((uint64_t)(uintptr_t)g) : "memory");
}
#define WAIT_ASYNC_6() asm volatile("s_wait_asynccnt 0x6" ::: "memory")
#define WAIT_ASYNC_0() asm volatile("s_wait_asynccnt 0x0" ::: "memory")

// ---------------------------------------------------------------------------
// Prepass 1: weights OIHW f32 -> [tap][co][ci] bf16 hi/lo planes
// ---------------------------------------------------------------------------
__global__ __launch_bounds__(256)
void prep_w(const float* __restrict__ w,
            unsigned short* __restrict__ wh, unsigned short* __restrict__ wl) {
    int n = blockIdx.x * 256 + threadIdx.x;        // over 9*192*192
    if (n >= 9 * COUT * CIN) return;
    int kpos = n / (COUT * CIN);
    int rem  = n - kpos * (COUT * CIN);
    int co = rem / CIN, ci = rem - co * CIN;
    float v = w[(co * CIN + ci) * 9 + kpos];
    unsigned short hi = f2bf(v);
    wh[n] = hi;
    wl[n] = f2bf(v - bf2f(hi));
}

// ---------------------------------------------------------------------------
// Prepass 2: x NCHW f32 -> NHWC bf16 hi/lo planes (LDS tile transpose)
// ---------------------------------------------------------------------------
__global__ __launch_bounds__(256)
void prep_x(const float* __restrict__ x,
            unsigned short* __restrict__ xh, unsigned short* __restrict__ xl) {
    __shared__ __align__(16) unsigned short tH[64 * RS];
    __shared__ __align__(16) unsigned short tL[64 * RS];

    int bid = blockIdx.x;                 // 4 wtiles * 6 cichunks * 1024 (b,h)
    int wt  = bid & 3;
    int cc  = (bid >> 2) % 6;
    int bh  = bid / 24;
    int h   = bh & (HD - 1);
    int b   = bh >> 8;
    int ci0 = cc * KC, w0 = wt * 64;

    int t = threadIdx.x;
    int wcol = t & 63, cir = t >> 6;      // read: coalesced along w
#pragma unroll
    for (int r = 0; r < 8; ++r) {
        int ci = r * 4 + cir;             // 0..31
        float v = x[((b * CIN + ci0 + ci) * HD + h) * WD + w0 + wcol];
        unsigned short hi = f2bf(v);
        tH[wcol * RS + ci] = hi;
        tL[wcol * RS + ci] = f2bf(v - bf2f(hi));
    }
    __syncthreads();
    int px = t >> 2, ce = (t & 3) * 8;    // write: 16B per thread, NHWC
    size_t g = ((size_t)((b * HD + h) * WD + w0 + px)) * CIN + ci0 + ce;
    *(u32x4*)&xh[g] = *(const u32x4*)&tH[px * RS + ce];
    *(u32x4*)&xl[g] = *(const u32x4*)&tL[px * RS + ce];
}

// ---------------------------------------------------------------------------
// Main kernel: async-staged, double-buffered implicit GEMM (64co x 128px)
// ---------------------------------------------------------------------------
__global__ __launch_bounds__(256)
void conv_main(const unsigned short* __restrict__ xh,
               const unsigned short* __restrict__ xl,
               const unsigned short* __restrict__ wh,
               const unsigned short* __restrict__ wl,
               const int*   __restrict__ mask,
               const float* __restrict__ bias,
               float*       __restrict__ out)
{
    __shared__ __align__(16) unsigned short sAh[2][MT * RS];
    __shared__ __align__(16) unsigned short sAl[2][MT * RS];
    __shared__ __align__(16) unsigned short sBh[2][NTM * RS];
    __shared__ __align__(16) unsigned short sBl[2][NTM * RS];

    const int bid = blockIdx.x;
    const int mt  = bid % 3;
    const int pt  = bid / 3;
    const int ow0 = (pt & 1) * NTM;          // 2 pixel tiles per row
    const int oh  = (pt >> 1) & (HD - 1);
    const int bb  = pt >> 9;                 // 512 pixel tiles per batch
    const int co0 = mt * MT;

    const int t    = threadIdx.x;
    const int lane = t & 31;
    const int wave = t >> 5;
    const int wm   = wave & 3;               // 4 waves along M (16 co)
    const int wn   = wave >> 2;              // 2 waves along N (64 px)
    const int half = lane >> 4;
    const int l16  = lane & 15;

    // staging coords: each thread moves one 16B chunk per row-pass per plane
    const int srow = t >> 2;                 // 0..63
    const int sce  = (t & 3) * 8;            // element offset {0,8,16,24}

    v8f acc[4] = {{}, {}, {}, {}};

    // issue the 6 async copies of K-step j into buffer `buf` (EXEC all-ones;
    // out-of-range activation rows use clamped addresses, fixed up later)
    auto issue_stage = [&](int j, int buf) {
        int kpos = j / 6, cc = j - kpos * 6;
        int kh = kpos / 3, kw = kpos - kh * 3;
        int ih = oh + kh - 1;
        int ci0 = cc * KC;
        // A: weights (2 asyncs)
        size_t gA = ((size_t)kpos * COUT + (co0 + srow)) * CIN + ci0 + sce;
        async_cp16((uint32_t)(uintptr_t)&sAh[buf][srow * RS + sce], wh + gA);
        async_cp16((uint32_t)(uintptr_t)&sAl[buf][srow * RS + sce], wl + gA);
        // B: activations (NHWC), 128 rows -> 2 row-passes (4 asyncs)
        int ihc = ih < 0 ? 0 : (ih >= HD ? HD - 1 : ih);
#pragma unroll
        for (int h2 = 0; h2 < 2; ++h2) {
            int row = srow + h2 * 64;
            int iw  = ow0 + row + kw - 1;
            int iwc = iw < 0 ? 0 : (iw >= WD ? WD - 1 : iw);
            size_t gB = ((size_t)((bb * HD + ihc) * WD + iwc)) * CIN + ci0 + sce;
            async_cp16((uint32_t)(uintptr_t)&sBh[buf][row * RS + sce], xh + gB);
            async_cp16((uint32_t)(uintptr_t)&sBl[buf][row * RS + sce], xl + gB);
        }
    };
    // zero-fill padding rows of step j (after its asyncs completed)
    auto fix_stage = [&](int j, int buf) {
        int kpos = j / 6;
        int kh = kpos / 3, kw = kpos - kh * 3;
        int ih = oh + kh - 1;
        bool ih_bad = (ih < 0 || ih >= HD);
#pragma unroll
        for (int h2 = 0; h2 < 2; ++h2) {
            int row = srow + h2 * 64;
            int iw  = ow0 + row + kw - 1;
            if (ih_bad || iw < 0 || iw >= WD) {
                u32x4 z = {0u, 0u, 0u, 0u};
                *(u32x4*)&sBh[buf][row * RS + sce] = z;
                *(u32x4*)&sBl[buf][row * RS + sce] = z;
            }
        }
    };

    // pipeline prologue
    issue_stage(0, 0);
    issue_stage(1, 1);
    WAIT_ASYNC_6();                      // stage 0 landed (in-order asyncs)
    fix_stage(0, 0);
    __syncthreads();

    for (int i = 0; i < KSTEPS; ++i) {
        const int buf = i & 1;

        // A fragments (16x32 bf16, ISA 7.12.2):
        // lanes 0-15: M=l16, K{0..7}@+0, K{16..23}@+16
        // lanes16-31: M=l16, K{8..15}@+8, K{24..31}@+24
        Frag ah, al;
        {
            int base = (wm * 16 + l16) * RS + half * 8;
            ah.q[0] = *(const u32x4*)&sAh[buf][base];
            ah.q[1] = *(const u32x4*)&sAh[buf][base + 16];
            al.q[0] = *(const u32x4*)&sAl[buf][base];
            al.q[1] = *(const u32x4*)&sAl[buf][base + 16];
        }
        // B fragments (32x16 bf16): lanes 0-15 N=l16 K{0..7},K{16..23}; lanes16-31 K{8..15},K{24..31}
        // 4 N-subtiles share the A fragments: 12 WMMAs per 20 ds_loads
#pragma unroll
        for (int s = 0; s < 4; ++s) {
            Frag bh, bl;
            int base = (wn * 64 + s * 16 + l16) * RS + half * 8;
            bh.q[0] = *(const u32x4*)&sBh[buf][base];
            bh.q[1] = *(const u32x4*)&sBh[buf][base + 16];
            bl.q[0] = *(const u32x4*)&sBl[buf][base];
            bl.q[1] = *(const u32x4*)&sBl[buf][base + 16];

            acc[s] = __builtin_amdgcn_wmma_f32_16x16x32_bf16(
                         false, ah.v, false, bh.v, (short)0, acc[s], false, false);
            acc[s] = __builtin_amdgcn_wmma_f32_16x16x32_bf16(
                         false, ah.v, false, bl.v, (short)0, acc[s], false, false);
            acc[s] = __builtin_amdgcn_wmma_f32_16x16x32_bf16(
                         false, al.v, false, bh.v, (short)0, acc[s], false, false);
        }

        if (i + 1 == KSTEPS) break;
        __syncthreads();                 // everyone done reading buf
        if (i + 2 < KSTEPS) {
            issue_stage(i + 2, buf);     // overlap copy with next WMMA step
            WAIT_ASYNC_6();              // stage i+1 landed
        } else {
            WAIT_ASYNC_0();
        }
        fix_stage(i + 1, (i + 1) & 1);
        __syncthreads();                 // stage i+1 visible to all waves
    }

    // epilogue: bias + 3x3 mask validity + store (C layout: M=r+8*half, N=l16)
    const int ow_base = ow0 + wn * 64 + l16;
    float vv[4];
#pragma unroll
    for (int s = 0; s < 4; ++s)
        vv[s] = valid_at(mask, bb, oh, ow_base + s * 16);

#pragma unroll
    for (int r = 0; r < 8; ++r) {
        int co = co0 + wm * 16 + half * 8 + r;
        float bv = bias[co];
        int o = ((bb * COUT + co) * HD + oh) * WD + ow_base;
#pragma unroll
        for (int s = 0; s < 4; ++s)
            out[o + s * 16] = (acc[s][r] + bv) * vv[s];
    }
}

// ---------------------------------------------------------------------------
// Fallback (small ws): self-contained kernel (in-loop conversion, 64co x 64px)
// ---------------------------------------------------------------------------
__global__ __launch_bounds__(256)
void conv_fallback(const float* __restrict__ x, const int* __restrict__ mask,
                   const float* __restrict__ w, const float* __restrict__ bias,
                   float* __restrict__ out)
{
    __shared__ __align__(16) unsigned short sAh[MT * RS], sAl[MT * RS];
    __shared__ __align__(16) unsigned short sBh[NT * RS], sBl[NT * RS];

    const int bid = blockIdx.x;
    const int mt  = bid % 3;
    const int pt  = bid / 3;
    const int ow0 = (pt & 3) * NT;
    const int oh  = (pt >> 2) & (HD - 1);
    const int bb  = pt >> 10;
    const int co0 = mt * MT;

    const int t    = threadIdx.x;
    const int lane = t & 31;
    const int wave = t >> 5;
    const int wm   = wave & 3;
    const int wn   = wave >> 2;
    const int half = lane >> 4;
    const int l16  = lane & 15;

    v8f acc0 = {}, acc1 = {};

    for (int kpos = 0; kpos < 9; ++kpos) {
        const int kh = kpos / 3, kw = kpos % 3;
        const int ih = oh + kh - 1;
        const bool ih_ok = (ih >= 0) && (ih < HD);
        for (int cc = 0; cc < CIN / KC; ++cc) {
            const int ci0 = cc * KC;
            __syncthreads();
#pragma unroll
            for (int r = 0; r < 8; ++r) {
                int idx = r * 256 + t;
                int co = idx >> 5, ci = idx & 31;
                float wv = w[((co0 + co) * CIN + (ci0 + ci)) * 9 + kpos];
                unsigned short hi = f2bf(wv);
                sAh[co * RS + ci] = hi;
                sAl[co * RS + ci] = f2bf(wv - bf2f(hi));
            }
#pragma unroll
            for (int r = 0; r < 8; ++r) {
                int idx = r * 256 + t;
                int px = idx & 63, ci = idx >> 6;
                int iw = ow0 + px + kw - 1;
                float xv = 0.0f;
                if (ih_ok && iw >= 0 && iw < WD)
                    xv = x[((bb * CIN + (ci0 + ci)) * HD + ih) * WD + iw];
                unsigned short hi = f2bf(xv);
                sBh[px * RS + ci] = hi;
                sBl[px * RS + ci] = f2bf(xv - bf2f(hi));
            }
            __syncthreads();

            Frag ah, al, bh0, bl0, bh1, bl1;
            {
                int base = (wm * 16 + l16) * RS + half * 8;
                ah.q[0] = *(const u32x4*)&sAh[base];
                ah.q[1] = *(const u32x4*)&sAh[base + 16];
                al.q[0] = *(const u32x4*)&sAl[base];
                al.q[1] = *(const u32x4*)&sAl[base + 16];
            }
            {
                int base0 = (wn * 32 + l16) * RS + half * 8;
                bh0.q[0] = *(const u32x4*)&sBh[base0];
                bh0.q[1] = *(const u32x4*)&sBh[base0 + 16];
                bl0.q[0] = *(const u32x4*)&sBl[base0];
                bl0.q[1] = *(const u32x4*)&sBl[base0 + 16];
                int base1 = base0 + 16 * RS;
                bh1.q[0] = *(const u32x4*)&sBh[base1];
                bh1.q[1] = *(const u32x4*)&sBh[base1 + 16];
                bl1.q[0] = *(const u32x4*)&sBl[base1];
                bl1.q[1] = *(const u32x4*)&sBl[base1 + 16];
            }
            acc0 = __builtin_amdgcn_wmma_f32_16x16x32_bf16(
                       false, ah.v, false, bh0.v, (short)0, acc0, false, false);
            acc0 = __builtin_amdgcn_wmma_f32_16x16x32_bf16(
                       false, ah.v, false, bl0.v, (short)0, acc0, false, false);
            acc0 = __builtin_amdgcn_wmma_f32_16x16x32_bf16(
                       false, al.v, false, bh0.v, (short)0, acc0, false, false);
            acc1 = __builtin_amdgcn_wmma_f32_16x16x32_bf16(
                       false, ah.v, false, bh1.v, (short)0, acc1, false, false);
            acc1 = __builtin_amdgcn_wmma_f32_16x16x32_bf16(
                       false, ah.v, false, bl1.v, (short)0, acc1, false, false);
            acc1 = __builtin_amdgcn_wmma_f32_16x16x32_bf16(
                       false, al.v, false, bh1.v, (short)0, acc1, false, false);
        }
    }

    const int ow_a = ow0 + wn * 32 + l16;
    const int ow_b = ow_a + 16;
    const float v0 = valid_at(mask, bb, oh, ow_a);
    const float v1 = valid_at(mask, bb, oh, ow_b);
#pragma unroll
    for (int r = 0; r < 8; ++r) {
        int co = co0 + wm * 16 + half * 8 + r;
        float bv = bias[co];
        int o = ((bb * COUT + co) * HD + oh) * WD + ow_a;
        out[o]      = (acc0[r] + bv) * v0;
        out[o + 16] = (acc1[r] + bv) * v1;
    }
}

extern "C" void kernel_launch(void* const* d_in, const int* in_sizes, int n_in,
                              void* d_out, int out_size, void* d_ws, size_t ws_size,
                              hipStream_t stream) {
    const float* x    = (const float*)d_in[0];
    const int*   mask = (const int*)  d_in[1];
    const float* w    = (const float*)d_in[2];
    const float* bias = (const float*)d_in[3];
    float*       out  = (float*)d_out;

    dim3 block(256);

    if (ws_size >= WS_NEED) {
        unsigned short* wh = (unsigned short*)((char*)d_ws + WH_OFF);
        unsigned short* wl = (unsigned short*)((char*)d_ws + WL_OFF);
        unsigned short* xh = (unsigned short*)((char*)d_ws + XH_OFF);
        unsigned short* xl = (unsigned short*)((char*)d_ws + XL_OFF);

        prep_w<<<dim3((9 * COUT * CIN + 255) / 256), block, 0, stream>>>(w, wh, wl);
        prep_x<<<dim3(4 * 6 * BN * HD), block, 0, stream>>>(x, xh, xl);
        conv_main<<<dim3(3 * BN * HD * (WD / NTM)), block, 0, stream>>>(
            xh, xl, wh, wl, mask, bias, out);
    } else {
        conv_fallback<<<dim3(3 * BN * HD * (WD / NT)), block, 0, stream>>>(
            x, mask, w, bias, out);
    }
}
